// NTXentLoss_15023795601702
// MI455X (gfx1250) — compile-verified, matrix operand order checked by reading, not run on hardware
//
#include <hip/hip_runtime.h>
#include <hip/hip_bf16.h>

typedef __attribute__((ext_vector_type(16))) _Float16 v16h;
typedef __attribute__((ext_vector_type(8)))  _Float16 v8h;
typedef __attribute__((ext_vector_type(8)))  float    v8f;

#define SPLITS 8
#define NTX_D 256
#define INV_T 14.285714285714286f   // 1 / 0.07

// ---------------------------------------------------------------------------
// Kernel 1: row L2-normalize, fp32 -> f16.  One 256-thread block per row.
// ---------------------------------------------------------------------------
__global__ __launch_bounds__(256) void ntx_normalize_kernel(
    const float* __restrict__ x, _Float16* __restrict__ fh, int D)
{
    const int row = blockIdx.x;
    const int t   = threadIdx.x;          // t in [0, 256) == D
    const float v = x[(size_t)row * D + t];

    float ss = v * v;
    #pragma unroll
    for (int m = 16; m > 0; m >>= 1)
        ss += __shfl_xor(ss, m, 32);

    __shared__ float ws[8];
    if ((t & 31) == 0) ws[t >> 5] = ss;
    __syncthreads();

    float tot = 0.f;
    #pragma unroll
    for (int i = 0; i < 8; ++i) tot += ws[i];

    const float scale = 1.0f / fmaxf(sqrtf(tot), 1e-8f);
    fh[(size_t)row * D + t] = (_Float16)(v * scale);
}

// ---------------------------------------------------------------------------
// Fragment loader: 16-bit A-matrix 16x32 per-lane layout (CDNA5 ISA 7.12.2).
// Lanes 0-15 hold row m=lane with K {0..7, 16..23}; lanes 16-31 hold row
// m=lane-16 with K {8..15, 24..31}.  Two 16-byte contiguous loads per lane.
// ---------------------------------------------------------------------------
static __device__ inline v16h ntx_load_frag(const _Float16* __restrict__ rowp, int kb)
{
    v8h lo = *(const v8h*)(rowp + kb);
    v8h hi = *(const v8h*)(rowp + kb + 16);
    return __builtin_shufflevector(lo, hi, 0, 1, 2, 3, 4, 5, 6, 7,
                                            8, 9, 10, 11, 12, 13, 14, 15);
}

// Load all 8 K-step B fragments of one 16-column tile into registers.
static __device__ inline void ntx_load_btile(const _Float16* __restrict__ f,
                                             int colBase, int m16, int koff,
                                             v16h* __restrict__ dst)
{
    const _Float16* brow = f + (size_t)(colBase + m16) * NTX_D;
    #pragma unroll
    for (int ks = 0; ks < 8; ++ks)
        dst[ks] = ntx_load_frag(brow, ks * 32 + koff);
}

// 8 back-to-back WMMAs over K=256, then branchless bounded-softexp epilogue:
// logits z = c/T are bounded by 1/T, so accumulate exp((c-1)/T) with no
// running max, no masking branch (diagonal corrected in the finalize pass).
static __device__ inline void ntx_accum_tile(const v16h* __restrict__ afrag,
                                             const v16h* __restrict__ bfrag,
                                             float* __restrict__ runSum)
{
    v8f c = {};
    #pragma unroll
    for (int ks = 0; ks < 8; ++ks)
        c = __builtin_amdgcn_wmma_f32_16x16x32_f16(
                false, afrag[ks], false, bfrag[ks],
                (short)0, c, false, false);
    #pragma unroll
    for (int g = 0; g < 8; ++g)
        runSum[g] += __expf((c[g] - 1.0f) * INV_T);
}

// ---------------------------------------------------------------------------
// Kernel 2: fused WMMA GEMM + bounded softexp accumulation.
// Block = 256 threads = 8 waves; each wave owns 16 rows, A fragments hoisted
// to registers (64 VGPRs); B tiles double-buffered in registers.
// grid = (N/128, SPLITS); partials[row*SPLITS+split] = sum exp((sim-1)/T).
// ---------------------------------------------------------------------------
__global__ __launch_bounds__(256) void ntx_simlse_kernel(
    const _Float16* __restrict__ f,
    float* __restrict__ partials,
    int N)
{
    const int lane    = threadIdx.x & 31;
    const int wave    = threadIdx.x >> 5;
    const int rowBase = blockIdx.x * 128 + wave * 16;
    const int colW    = N / SPLITS;                 // columns per split
    const int col0    = blockIdx.y * colW;
    const int nTiles  = colW >> 4;                  // 16-col tiles (even)

    const int m16   = lane & 15;                    // row index for frag loads
    const int koff  = (lane >> 4) * 8;              // K sub-chunk per lane half
    const int rHalf = (lane >> 4) << 3;             // C-frag row offset (0 or 8)

    // --- hoist all A fragments (16 rows x K=256) into registers ---
    const _Float16* arow = f + (size_t)(rowBase + m16) * NTX_D;
    v16h afrag[8];
    #pragma unroll
    for (int ks = 0; ks < 8; ++ks)
        afrag[ks] = ntx_load_frag(arow, ks * 32 + koff);

    float runSum[8];
    #pragma unroll
    for (int g = 0; g < 8; ++g) runSum[g] = 0.f;

    // --- double-buffered column-tile loop ---
    v16h b0[8], b1[8];
    ntx_load_btile(f, col0, m16, koff, b0);
    for (int ct = 0; ct < nTiles; ct += 2) {
        ntx_load_btile(f, col0 + (ct + 1) * 16, m16, koff, b1);
        ntx_accum_tile(afrag, b0, runSum);
        if (ct + 2 < nTiles)
            ntx_load_btile(f, col0 + (ct + 2) * 16, m16, koff, b0);
        ntx_accum_tile(afrag, b1, runSum);
    }

    // --- sum the 16 lanes of each half (they cover disjoint columns) ---
    #pragma unroll
    for (int g = 0; g < 8; ++g) {
        float s = runSum[g];
        #pragma unroll
        for (int msk = 1; msk < 16; msk <<= 1)
            s += __shfl_xor(s, msk, 32);
        if ((lane & 15) == 0) {
            const int grow = rowBase + rHalf + g;
            partials[(size_t)grow * SPLITS + blockIdx.y] = s;
        }
    }
}

// ---------------------------------------------------------------------------
// Kernel 3: per-row finalize.  One wave per row: recompute self-dot (diagonal
// correction) and positive-pair dot, merge split partials, emit nll[row].
// ---------------------------------------------------------------------------
__global__ __launch_bounds__(256) void ntx_rownll_kernel(
    const _Float16* __restrict__ fh,
    const float* __restrict__ partials,
    float* __restrict__ nll, int N)
{
    const int row  = blockIdx.x * 8 + (threadIdx.x >> 5);
    const int lane = threadIdx.x & 31;
    const _Float16* fi = fh + (size_t)row * NTX_D;
    const _Float16* fp = fh + (size_t)((row + (N >> 1)) & (N - 1)) * NTX_D;

    v8h a = *(const v8h*)(fi + lane * 8);
    v8h b = *(const v8h*)(fp + lane * 8);
    float selfs = 0.f, poss = 0.f;
    #pragma unroll
    for (int i = 0; i < 8; ++i) {
        const float av = (float)a[i];
        selfs += av * av;
        poss  += av * (float)b[i];
    }
    #pragma unroll
    for (int m = 16; m > 0; m >>= 1) {
        selfs += __shfl_xor(selfs, m, 32);
        poss  += __shfl_xor(poss,  m, 32);
    }

    if (lane == 0) {
        float s = 0.f;
        #pragma unroll
        for (int sp = 0; sp < SPLITS; ++sp)
            s += partials[(size_t)row * SPLITS + sp];
        s -= __expf((selfs - 1.0f) * INV_T);       // remove diagonal term
        // lse = 1/T + log(sum exp(z - 1/T));  nll = lse - z_pos
        nll[row] = INV_T + __logf(s) - poss * INV_T;
    }
}

// ---------------------------------------------------------------------------
// Kernel 4: mean over N rows.  Single 256-thread block.
// ---------------------------------------------------------------------------
__global__ __launch_bounds__(256) void ntx_mean_kernel(
    const float* __restrict__ nll, float* __restrict__ out, int N)
{
    float local = 0.f;
    for (int i = threadIdx.x; i < N; i += 256)
        local += nll[i];

    __shared__ float red[256];
    red[threadIdx.x] = local;
    __syncthreads();
    #pragma unroll
    for (int off = 128; off > 0; off >>= 1) {
        if (threadIdx.x < off) red[threadIdx.x] += red[threadIdx.x + off];
        __syncthreads();
    }
    if (threadIdx.x == 0) out[0] = red[0] / (float)N;
}

// ---------------------------------------------------------------------------
extern "C" void kernel_launch(void* const* d_in, const int* in_sizes, int n_in,
                              void* d_out, int out_size, void* d_ws, size_t ws_size,
                              hipStream_t stream)
{
    const float* feat = (const float*)d_in[0];
    const int D = NTX_D;
    const int N = in_sizes[0] / D;                 // 8192

    // workspace layout: [f16 normalized | float partials | float nll]
    _Float16* fh   = (_Float16*)d_ws;
    char*     p1   = (char*)d_ws + (size_t)N * D * sizeof(_Float16);
    float*    part = (float*)p1;
    float*    nll  = (float*)(p1 + (size_t)N * SPLITS * sizeof(float));

    ntx_normalize_kernel<<<N, 256, 0, stream>>>(feat, fh, D);

    dim3 grid(N / 128, SPLITS);
    ntx_simlse_kernel<<<grid, 256, 0, stream>>>(fh, part, N);

    ntx_rownll_kernel<<<N / 8, 256, 0, stream>>>(fh, part, nll, N);

    ntx_mean_kernel<<<1, 256, 0, stream>>>(nll, (float*)d_out, N);
}
